// SlidingWindowAttention_46892452938097
// MI455X (gfx1250) — matrix operand order repaired and checked by reading, not compiled, not run on hardware
//
#include <hip/hip_runtime.h>

// ---------------- problem constants ----------------
#define BATCH    2
#define SEQ_T    4096
#define N_EMBD   2048
#define N_HEAD   32
#define N_KVH    8
#define HEAD_DIM 64
#define WINDOW   1024

typedef __attribute__((ext_vector_type(16))) __bf16        v16bf;
typedef __attribute__((ext_vector_type(8)))  float         v8f;
typedef __attribute__((ext_vector_type(4)))  unsigned int  v4u;

union Frag { v16bf bf; v4u u[2]; };

__device__ __forceinline__ v8f wmma_bf16(Frag a, Frag b, v8f c) {
  // D = A(16x32 bf16) * B(32x16 bf16) + C(16x16 f32)
  return __builtin_amdgcn_wmma_f32_16x16x32_bf16(false, a.bf, false, b.bf,
                                                 (short)0, c, false, false);
}

// max-reduce across the 16-lane C-matrix group via ds_swizzle (group-of-32 xor mode:
// offset = xor_mask<<10 | and_mask 0x1f); xor masks 1,2,4,8 stay inside each half.
__device__ __forceinline__ float swz_max16(float x) {
  x = fmaxf(x, __int_as_float(__builtin_amdgcn_ds_swizzle(__float_as_int(x), 0x041f)));
  x = fmaxf(x, __int_as_float(__builtin_amdgcn_ds_swizzle(__float_as_int(x), 0x081f)));
  x = fmaxf(x, __int_as_float(__builtin_amdgcn_ds_swizzle(__float_as_int(x), 0x101f)));
  x = fmaxf(x, __int_as_float(__builtin_amdgcn_ds_swizzle(__float_as_int(x), 0x201f)));
  return x;
}

// ---------------- conversion kernels ----------------
__global__ void cvt_f32_bf16(const float* __restrict__ in, __bf16* __restrict__ out, size_t n) {
  size_t i = (size_t)blockIdx.x * blockDim.x + threadIdx.x;
  if (i < n) out[i] = (__bf16)in[i];
}

// W [K,N] f32 row-major -> Wt [N,K] bf16 row-major (K-major for B fragments)
__global__ void cvt_tr_bf16(const float* __restrict__ W, __bf16* __restrict__ Wt, int K, int N) {
  size_t i = (size_t)blockIdx.x * blockDim.x + threadIdx.x;
  if (i >= (size_t)K * N) return;
  int k = (int)(i / N), n = (int)(i % N);
  Wt[(size_t)n * K + k] = (__bf16)W[i];
}

// ---------------- WMMA GEMM: C[M,N] = A[M,K](bf16,row) * Bt[N,K](bf16,K-major) ----------------
// block = 8 waves -> 64x128 tile; wave tile = 16 rows x 64 cols (4 accumulators)
template <bool F32OUT>
__global__ __launch_bounds__(256) void gemm_wmma_kernel(
    const __bf16* __restrict__ A, const __bf16* __restrict__ Bt,
    void* __restrict__ Cp, int M, int N, int K)
{
  const int lane = threadIdx.x & 31;
  const int wave = threadIdx.x >> 5;
  const int g  = lane >> 4;      // half-wave group
  const int lo = lane & 15;

  const int m0 = blockIdx.x * 64  + (wave >> 1) * 16;
  const int n0 = blockIdx.y * 128 + (wave & 1) * 64;

  v8f acc[4] = {v8f{}, v8f{}, v8f{}, v8f{}};
  const __bf16* arow = A + (size_t)(m0 + lo) * K;
  const __bf16* brow = Bt + (size_t)(n0 + lo) * K;

  for (int kk = 0; kk < K; kk += 32) {
    // prefetch ~4 K-steps ahead (speculative; past-end fetches are harmless)
    __builtin_prefetch(arow + kk + 128, 0, 1);
    __builtin_prefetch(brow + kk + 128, 0, 1);

    Frag a;
    const v4u* ap = (const v4u*)(arow + kk + 16 * g);
    a.u[0] = ap[0];      // k = 16g .. 16g+7
    a.u[1] = ap[1];      // k = 16g+8 .. 16g+15
#pragma unroll
    for (int c = 0; c < 4; ++c) {
      const v4u* bp = (const v4u*)(Bt + (size_t)(n0 + c * 16 + lo) * K + kk + 16 * g);
      Frag b; b.u[0] = bp[0]; b.u[1] = bp[1];   // k = 16g .. 16g+15
      acc[c] = wmma_bf16(a, b, acc[c]);
    }
  }
#pragma unroll
  for (int c = 0; c < 4; ++c)
#pragma unroll
    for (int r = 0; r < 8; ++r) {
      size_t idx = (size_t)(m0 + r + 8 * g) * N + n0 + c * 16 + lo;
      if (F32OUT) ((float*)Cp)[idx] = acc[c][r];
      else        ((__bf16*)Cp)[idx] = (__bf16)acc[c][r];
    }
}

// ---------------- RoPE + layout kernels ----------------
// ln(10000)/32
#define ROPE_LNB_OVER 0.28782313662425572f

// Qlin [B*T, 2048] bf16 -> Qr [B,H,T,64] bf16 (roped)
__global__ void rope_q_kernel(const __bf16* __restrict__ Qlin, __bf16* __restrict__ Qr) {
  size_t i = (size_t)blockIdx.x * blockDim.x + threadIdx.x;  // B*T*H*32
  int j = (int)(i & 31);
  size_t r = i >> 5;
  int h = (int)(r % N_HEAD); r /= N_HEAD;
  int t = (int)(r % SEQ_T);
  int b = (int)(r / SEQ_T);
  float ang = (float)t * __expf(-(float)j * ROPE_LNB_OVER);
  float c = __cosf(ang), s = __sinf(ang);
  const __bf16* src = Qlin + ((size_t)b * SEQ_T + t) * N_EMBD + h * HEAD_DIM;
  float x1 = (float)src[j], x2 = (float)src[j + 32];
  __bf16* dst = Qr + (((size_t)(b * N_HEAD + h) * SEQ_T) + t) * HEAD_DIM;
  dst[j]      = (__bf16)(x1 * c - x2 * s);
  dst[j + 32] = (__bf16)(x2 * c + x1 * s);
}

// KVlin [B*T, 1024] bf16 -> Kr [B,KV,T,64] (roped), Vt [B,KV,64,T] (transposed copy)
__global__ void rope_kv_kernel(const __bf16* __restrict__ KVlin,
                               __bf16* __restrict__ Kr, __bf16* __restrict__ Vt) {
  size_t i = (size_t)blockIdx.x * blockDim.x + threadIdx.x;  // B*T*KV*32
  int j = (int)(i & 31);
  size_t r = i >> 5;
  int kvh = (int)(r % N_KVH); r /= N_KVH;
  int t = (int)(r % SEQ_T);
  int b = (int)(r / SEQ_T);
  float ang = (float)t * __expf(-(float)j * ROPE_LNB_OVER);
  float c = __cosf(ang), s = __sinf(ang);
  const __bf16* row = KVlin + ((size_t)b * SEQ_T + t) * (2 * N_KVH * HEAD_DIM);
  float k1 = (float)row[kvh * HEAD_DIM + j];
  float k2 = (float)row[kvh * HEAD_DIM + j + 32];
  float v1 = (float)row[N_KVH * HEAD_DIM + kvh * HEAD_DIM + j];
  float v2 = (float)row[N_KVH * HEAD_DIM + kvh * HEAD_DIM + j + 32];
  __bf16* kd = Kr + (((size_t)(b * N_KVH + kvh) * SEQ_T) + t) * HEAD_DIM;
  kd[j]      = (__bf16)(k1 * c - k2 * s);
  kd[j + 32] = (__bf16)(k2 * c + k1 * s);
  __bf16* vbase = Vt + (size_t)(b * N_KVH + kvh) * HEAD_DIM * SEQ_T;
  vbase[(size_t)j * SEQ_T + t]        = (__bf16)v1;
  vbase[(size_t)(j + 32) * SEQ_T + t] = (__bf16)v2;
}

// ---------------- flash attention, sliding window ----------------
// One wave per 16-query tile of one (b,h); 8 waves / block. 32-key tiles.
// Softmax denominator computed by the matrix core: l_tile = P x ones(32x16).
__global__ __launch_bounds__(256) void attn_kernel(
    const __bf16* __restrict__ Q,   // [B,H,T,64]
    const __bf16* __restrict__ Kc,  // [B,KV,T,64]
    const __bf16* __restrict__ Vt,  // [B,KV,64,T]
    __bf16* __restrict__ O)         // [B*T, 2048]
{
  __shared__ __align__(16) __bf16 pbuf[8][16][48];  // 96B row stride keeps b128 LDS loads aligned

  const int lane = threadIdx.x & 31;
  const int wave = threadIdx.x >> 5;
  const int g  = lane >> 4;
  const int lo = lane & 15;

  const int qtiles = SEQ_T / 16;
  int gw   = blockIdx.x * 8 + wave;
  int qt   = gw % qtiles;
  int rest = gw / qtiles;
  int h = rest % N_HEAD;
  int b = rest / N_HEAD;
  int kvh = h >> 2;            // N_REP = 4
  int q0 = qt * 16;

  // Q A-fragments for d in [0,32) and [32,64)
  const __bf16* qrow = Q + (((size_t)(b * N_HEAD + h) * SEQ_T) + q0 + lo) * HEAD_DIM;
  const v4u* qp = (const v4u*)(qrow + 16 * g);
  Frag aq0, aq1;
  aq0.u[0] = qp[0]; aq0.u[1] = qp[1];
  aq1.u[0] = qp[4]; aq1.u[1] = qp[5];

  // all-ones B fragment for matrix-core row sums
  Frag ones;
#pragma unroll
  for (int i = 0; i < 16; ++i) ones.bf[i] = (__bf16)1.0f;

  const __bf16* Kbase = Kc + (size_t)(b * N_KVH + kvh) * SEQ_T * HEAD_DIM;
  const __bf16* Vbase = Vt + (size_t)(b * N_KVH + kvh) * HEAD_DIM * SEQ_T;

  float ml[8];
  v8f o[4] = {v8f{}, v8f{}, v8f{}, v8f{}};
  v8f lacc = v8f{};
#pragma unroll
  for (int r = 0; r < 8; ++r) ml[r] = -1e30f;

  int kstart = q0 - (WINDOW - 1);
  if (kstart < 0) kstart = 0;
  kstart &= ~31;

  for (int kt = kstart; kt < q0 + 16; kt += 32) {
    // K^T B-fragments: contraction over head dim (contiguous in K layout)
    const v4u* kp0 = (const v4u*)(Kbase + (size_t)(kt + lo) * HEAD_DIM + 16 * g);
    const v4u* kp1 = (const v4u*)(Kbase + (size_t)(kt + 16 + lo) * HEAD_DIM + 16 * g);
    Frag b00, b01, b10, b11;
    b00.u[0] = kp0[0]; b00.u[1] = kp0[1];   // d = 16g..16g+15
    b01.u[0] = kp0[4]; b01.u[1] = kp0[5];   // d = 32+16g..
    b10.u[0] = kp1[0]; b10.u[1] = kp1[1];
    b11.u[0] = kp1[4]; b11.u[1] = kp1[5];

    v8f s0 = v8f{}, s1 = v8f{};
    s0 = wmma_bf16(aq0, b00, s0);
    s0 = wmma_bf16(aq1, b01, s0);
    s1 = wmma_bf16(aq0, b10, s1);
    s1 = wmma_bf16(aq1, b11, s1);

    // online softmax: rows live across the 16-lane groups; only the max needs shuffles
#pragma unroll
    for (int r = 0; r < 8; ++r) {
      int qpos = q0 + r + 8 * g;
      int k0 = kt + lo, k1 = kt + 16 + lo;
      bool ok0 = (k0 <= qpos) && (k0 > qpos - WINDOW);
      bool ok1 = (k1 <= qpos) && (k1 > qpos - WINDOW);
      float v0 = ok0 ? s0[r] * 0.125f : -1e30f;
      float v1 = ok1 ? s1[r] * 0.125f : -1e30f;
      float t = swz_max16(fmaxf(v0, v1));
      float mnew  = fmaxf(ml[r], t);
      float alpha = __expf(ml[r] - mnew);
      float e0 = ok0 ? __expf(v0 - mnew) : 0.f;
      float e1 = ok1 ? __expf(v1 - mnew) : 0.f;
      ml[r] = mnew;
      lacc[r] *= alpha;
#pragma unroll
      for (int c = 0; c < 4; ++c) o[c][r] *= alpha;
      pbuf[wave][r + 8 * g][lo]      = (__bf16)e0;
      pbuf[wave][r + 8 * g][16 + lo] = (__bf16)e1;
    }

    // per-wave LDS transpose of P: DS ops are in-order within a wave; make it explicit
    asm volatile("s_wait_dscnt 0" ::: "memory");

    Frag pa;
    const v4u* pp = (const v4u*)(&pbuf[wave][lo][16 * g]);
    pa.u[0] = pp[0];   // keys 16g..16g+7
    pa.u[1] = pp[1];   // keys 16g+8..16g+15

    // denominator: lacc += P x ones (row sum replicated across columns)
    lacc = wmma_bf16(pa, ones, lacc);

    // O += P(16x32) * V(32keys x 16d) from transposed V (contiguous along T)
#pragma unroll
    for (int c = 0; c < 4; ++c) {
      const v4u* vp = (const v4u*)(Vbase + (size_t)(c * 16 + lo) * SEQ_T + kt + 16 * g);
      Frag bv; bv.u[0] = vp[0]; bv.u[1] = vp[1];
      o[c] = wmma_bf16(pa, bv, o[c]);
    }
  }

  // normalize and store to [B*T, E] bf16 for the output projection
#pragma unroll
  for (int r = 0; r < 8; ++r) {
    float inv = 1.0f / lacc[r];
    size_t row = (size_t)b * SEQ_T + q0 + r + 8 * g;
#pragma unroll
    for (int c = 0; c < 4; ++c)
      O[row * N_EMBD + h * HEAD_DIM + c * 16 + lo] = (__bf16)(o[c][r] * inv);
  }
}

// ---------------- launcher ----------------
extern "C" void kernel_launch(void* const* d_in, const int* in_sizes, int n_in,
                              void* d_out, int out_size, void* d_ws, size_t ws_size,
                              hipStream_t stream) {
  (void)in_sizes; (void)n_in; (void)out_size; (void)ws_size;
  const float* x   = (const float*)d_in[0];
  const float* Wq  = (const float*)d_in[1];
  const float* Wkv = (const float*)d_in[2];
  const float* Wo  = (const float*)d_in[3];
  float* out = (float*)d_out;

  const size_t M   = (size_t)BATCH * SEQ_T;          // 8192
  const size_t NX  = M * N_EMBD;                     // x elements
  const size_t NKV = M * (2 * N_KVH * HEAD_DIM);     // kv elements
  const size_t NKVHD = (size_t)BATCH * N_KVH * SEQ_T * HEAD_DIM;

  size_t off = 0;
  auto wsalloc = [&](size_t bytes) -> void* {
    void* p = (char*)d_ws + off;
    off += (bytes + 255) & ~(size_t)255;
    return p;
  };
  __bf16* xb    = (__bf16*)wsalloc(NX * 2);
  __bf16* Wqt   = (__bf16*)wsalloc((size_t)N_EMBD * N_EMBD * 2);
  __bf16* Wkvt  = (__bf16*)wsalloc((size_t)N_EMBD * 1024 * 2);
  __bf16* Wot   = (__bf16*)wsalloc((size_t)N_EMBD * N_EMBD * 2);
  __bf16* Qlin  = (__bf16*)wsalloc(NX * 2);
  __bf16* KVlin = (__bf16*)wsalloc(NKV * 2);
  __bf16* Qr    = (__bf16*)wsalloc(NX * 2);
  __bf16* Kr    = (__bf16*)wsalloc(NKVHD * 2);
  __bf16* Vtr   = (__bf16*)wsalloc(NKVHD * 2);
  __bf16* AttnO = (__bf16*)wsalloc(NX * 2);

  // 1. convert activations and weights (weights transposed to K-major)
  cvt_f32_bf16<<<(unsigned)((NX + 255) / 256), 256, 0, stream>>>(x, xb, NX);
  cvt_tr_bf16<<<(unsigned)(((size_t)N_EMBD * N_EMBD + 255) / 256), 256, 0, stream>>>(Wq,  Wqt,  N_EMBD, N_EMBD);
  cvt_tr_bf16<<<(unsigned)(((size_t)N_EMBD * 1024   + 255) / 256), 256, 0, stream>>>(Wkv, Wkvt, N_EMBD, 1024);
  cvt_tr_bf16<<<(unsigned)(((size_t)N_EMBD * N_EMBD + 255) / 256), 256, 0, stream>>>(Wo,  Wot,  N_EMBD, N_EMBD);

  // 2. projections (bf16 WMMA, f32 accumulate, bf16 store)
  dim3 gq((unsigned)(M / 64), N_EMBD / 128);
  gemm_wmma_kernel<false><<<gq, 256, 0, stream>>>(xb, Wqt, Qlin, (int)M, N_EMBD, N_EMBD);
  dim3 gkv((unsigned)(M / 64), 1024 / 128);
  gemm_wmma_kernel<false><<<gkv, 256, 0, stream>>>(xb, Wkvt, KVlin, (int)M, 1024, N_EMBD);

  // 3. RoPE + layout for attention
  rope_q_kernel<<<(unsigned)(((size_t)BATCH * SEQ_T * N_HEAD * 32) / 256), 256, 0, stream>>>(Qlin, Qr);
  rope_kv_kernel<<<(unsigned)(((size_t)BATCH * SEQ_T * N_KVH * 32) / 256), 256, 0, stream>>>(KVlin, Kr, Vtr);

  // 4. sliding-window flash attention (WMMA QK^T, PV, and row-sum)
  unsigned attn_blocks = (unsigned)((size_t)BATCH * N_HEAD * (SEQ_T / 16) / 8);
  attn_kernel<<<attn_blocks, 256, 0, stream>>>(Qr, Kr, Vtr, AttnO);

  // 5. output projection, f32 store to d_out
  gemm_wmma_kernel<true><<<gq, 256, 0, stream>>>(AttnO, Wot, out, (int)M, N_EMBD, N_EMBD);
}
